// SeededSelfAttention_1486058684978
// MI455X (gfx1250) — matrix operand order repaired and checked by reading, not compiled, not run on hardware
//
#include <hip/hip_runtime.h>
#include <hip/hip_bf16.h>

#define DEV __device__ __forceinline__

typedef __attribute__((ext_vector_type(16))) __bf16 v16bf;
typedef __attribute__((ext_vector_type(8)))  float  v8f;
typedef __attribute__((address_space(3))) __bf16 lds_bf16;

union Pack16 { __bf16 h[16]; uint4 u[2]; };
union FragU  { uint4 u[2]; v16bf v; };

// A-operand fragment (16x32 bf16): lane (hi = lane>>4) holds K=[8*hi..+7] in
// VGPR0-3 and K=[16+8*hi..+7] in VGPR4-7 (two 16B contiguous loads).
DEV v16bf frag_a(const __bf16* row_k, int hi8) {
  FragU f;
  f.u[0] = *(const uint4*)(row_k + hi8);
  f.u[1] = *(const uint4*)(row_k + 16 + hi8);
  return f.v;
}
// B-operand fragment (32x16 bf16): lane holds column n = lane&15,
// K=[8*hi..+7] in VGPR0-3 and K=[16+8*hi..+7] in VGPR4-7.
DEV v16bf frag_b(const __bf16* col_k, int hi16) {
  FragU f;
  f.u[0] = *(const uint4*)(col_k + hi16 / 2);
  f.u[1] = *(const uint4*)(col_k + 16 + hi16 / 2);
  return f.v;
}
DEV v8f wmma_bf16(v16bf a, v16bf b, v8f c) {
  return __builtin_amdgcn_wmma_f32_16x16x32_bf16(false, a, false, b, (short)0, c, false, false);
}
DEV void cvt16(Pack16& t, const float4* s) {
#pragma unroll
  for (int i = 0; i < 4; i++) {
    float4 f = s[i];
    t.h[4*i+0] = (__bf16)f.x; t.h[4*i+1] = (__bf16)f.y;
    t.h[4*i+2] = (__bf16)f.z; t.h[4*i+3] = (__bf16)f.w;
  }
}

// ---------------------------------------------------------------------------
// K0: x fp32 [B,N,D] -> xb bf16 [B,N,D] and xt bf16 [B,D,N] (LDS transpose)
// ---------------------------------------------------------------------------
__global__ void k_convert(const float* __restrict__ x,
                          __bf16* __restrict__ xb, __bf16* __restrict__ xt) {
  __shared__ __bf16 tile[64 * 72];          // [d_local][n_local], padded stride
  const int N = 2048, D = 1024;
  int b = blockIdx.z, n0 = blockIdx.x * 64, d0 = blockIdx.y * 64;
  int tid = threadIdx.x;
  int r = tid >> 2, c0 = (tid & 3) * 16;    // r: n_local, c0: d_local start
  const float* src = x + ((size_t)b * N + n0 + r) * D + d0 + c0;
  Pack16 t;
  cvt16(t, (const float4*)src);
  __bf16* dxb = xb + ((size_t)b * N + n0 + r) * D + d0 + c0;
  ((uint4*)dxb)[0] = t.u[0]; ((uint4*)dxb)[1] = t.u[1];
#pragma unroll
  for (int i = 0; i < 16; i++) tile[(c0 + i) * 72 + r] = t.h[i];
  __syncthreads();
  int dr = tid >> 2, nc0 = (tid & 3) * 16;
  Pack16 o;
  o.u[0] = *(const uint4*)&tile[dr * 72 + nc0];
  o.u[1] = *(const uint4*)&tile[dr * 72 + nc0 + 8];
  __bf16* dxt = xt + ((size_t)b * D + d0 + dr) * N + n0 + nc0;
  ((uint4*)dxt)[0] = o.u[0]; ((uint4*)dxt)[1] = o.u[1];
}

// ---------------------------------------------------------------------------
// K1: Q' = (seed @ Wk^T) * (1/sqrt(P))  -> bf16 [512,1024]
// Software-pipelined: next chunk's global loads issued before WMMA block.
// ---------------------------------------------------------------------------
__global__ void k_qproj(const float* __restrict__ seed, const float* __restrict__ Wk,
                        __bf16* __restrict__ Qb) {
  __shared__ __bf16 lA[128 * 40];
  __shared__ __bf16 lB[128 * 40];
  int d0 = blockIdx.x * 128, s0 = blockIdx.y * 128;
  int tid = threadIdx.x, w = tid >> 5, lane = tid & 31;
  int hi8 = (lane >> 4) * 8, hi16 = (lane >> 4) * 16, lr = lane & 15;
  int wm = (w >> 2) * 64, wn = (w & 3) * 32;
  v8f acc[4][2] = {};
  int srow = tid >> 1, sc0 = (tid & 1) * 16;
  const float* pa = seed + (size_t)(s0 + srow) * 1024 + sc0;
  const float* pb = Wk   + (size_t)(d0 + srow) * 1024 + sc0;
  float4 ra[4], rb[4];
#pragma unroll
  for (int i = 0; i < 4; i++) { ra[i] = ((const float4*)pa)[i]; rb[i] = ((const float4*)pb)[i]; }
  for (int k0 = 0; k0 < 1024; k0 += 32) {
    Pack16 t, u;
    cvt16(t, ra); cvt16(u, rb);
    uint4* da = (uint4*)&lA[srow * 40 + sc0]; da[0] = t.u[0]; da[1] = t.u[1];
    uint4* db = (uint4*)&lB[srow * 40 + sc0]; db[0] = u.u[0]; db[1] = u.u[1];
    __syncthreads();
    if (k0 + 32 < 1024) {                       // overlap next loads with WMMA
      const float* na = pa + k0 + 32;
      const float* nb = pb + k0 + 32;
#pragma unroll
      for (int i = 0; i < 4; i++) { ra[i] = ((const float4*)na)[i]; rb[i] = ((const float4*)nb)[i]; }
    }
    v16bf bfr[2];
#pragma unroll
    for (int nt = 0; nt < 2; nt++) bfr[nt] = frag_b(&lB[(wn + nt * 16 + lr) * 40], hi16);
#pragma unroll
    for (int mt = 0; mt < 4; mt++) {
      v16bf a = frag_a(&lA[(wm + mt * 16 + lr) * 40], hi8);
#pragma unroll
      for (int nt = 0; nt < 2; nt++) acc[mt][nt] = wmma_bf16(a, bfr[nt], acc[mt][nt]);
    }
    __syncthreads();
  }
  const float scale = 0.03125f;  // 1/sqrt(1024)
#pragma unroll
  for (int mt = 0; mt < 4; mt++)
#pragma unroll
    for (int nt = 0; nt < 2; nt++)
#pragma unroll
      for (int j = 0; j < 8; j++) {
        int sr = s0 + wm + mt * 16 + j + hi8;     // C-row = vgpr + 8*hi
        int dc = d0 + wn + nt * 16 + lr;
        Qb[(size_t)sr * 1024 + dc] = (__bf16)(acc[mt][nt][j] * scale);
      }
}

// ---------------------------------------------------------------------------
// K2: flash attention per (batch, 16-row s-tile):
//   T[b,s,:] = softmax(Q' @ xb[b]^T) @ xb[b]   (bk dropped: shift invariance)
// Q tile staged to LDS via GLOBAL_LOAD_ASYNC_TO_LDS (ASYNCcnt). Softmax is a
// two-level parallel reduction. 8 waves split n for logits, d for accumulation.
// ---------------------------------------------------------------------------
__global__ void k_attn(const __bf16* __restrict__ Qb, const __bf16* __restrict__ xb,
                       const __bf16* __restrict__ xt, __bf16* __restrict__ T) {
  const int N = 2048, D = 1024;
  __shared__ __bf16 Qt[16 * 1024];
  __shared__ float  Sb[16 * 256];
  __shared__ __bf16 Pb[16 * 256];
  __shared__ float  red[16 * 16];
  __shared__ float  mrow[16], lrow[16], rrow[16], mnew[16];
  int b = blockIdx.y, s0 = blockIdx.x * 16;
  int tid = threadIdx.x, w = tid >> 5, lane = tid & 31;
  int hi8 = (lane >> 4) * 8, hi16 = (lane >> 4) * 16, lr = lane & 15;
  const __bf16* xbp = xb + (size_t)b * N * D;
  const __bf16* xtp = xt + (size_t)b * D * N;
  {
    // Async DMA the 32KB Q tile into LDS (128B per thread, 8x b128 each).
    int row = tid >> 4, c0 = (tid & 15) * 64;
    const __bf16* g = Qb + (size_t)(s0 + row) * D + c0;
    unsigned loff = (unsigned)(unsigned long long)(lds_bf16*)&Qt[row * D + c0];
#pragma unroll
    for (int i = 0; i < 8; i++) {
      asm volatile("global_load_async_to_lds_b128 %0, %1, off"
                   :: "v"(loff + 16u * i), "v"(g + 8 * i) : "memory");
    }
    asm volatile("s_wait_asynccnt 0x0" ::: "memory");
  }
  if (tid < 16) { mrow[tid] = -3.0e38f; lrow[tid] = 0.f; }
  v8f acc[8] = {};
  int dbase = w * 128;
  __syncthreads();
  for (int n0 = 0; n0 < N; n0 += 256) {
    // --- logits chunk: this wave covers n = n0 + 32*w .. +31 ---
    v8f sacc[2] = {};
    int nw = n0 + w * 32;
    if (n0 + 256 < N) {
      __builtin_prefetch(xbp + (size_t)(nw + 256) * D, 0, 1);
      __builtin_prefetch(xtp + (size_t)(dbase + lr) * N + n0 + 256, 0, 1);
    }
    for (int k0 = 0; k0 < D; k0 += 32) {
      v16bf a = frag_a(&Qt[lr * D + k0], hi8);
#pragma unroll
      for (int nt = 0; nt < 2; nt++) {
        const __bf16* col = xbp + (size_t)(nw + nt * 16 + lr) * D + k0;
        sacc[nt] = wmma_bf16(a, frag_b(col, hi16), sacc[nt]);
      }
    }
#pragma unroll
    for (int nt = 0; nt < 2; nt++)
#pragma unroll
      for (int j = 0; j < 8; j++)
        Sb[(j + hi8) * 256 + (w * 32 + nt * 16 + lr)] = sacc[nt][j];
    __syncthreads();
    // --- online softmax: parallel max over 16-elem segments ---
    {
      int r = tid >> 4, seg = (tid & 15) * 16;
      float lm = -3.0e38f;
#pragma unroll
      for (int i = 0; i < 16; i++) lm = fmaxf(lm, Sb[r * 256 + seg + i]);
      red[(tid >> 4) * 16 + (tid & 15)] = lm;
    }
    __syncthreads();
    if (tid < 16) {
      float mo = mrow[tid], cm = mo;
#pragma unroll
      for (int i = 0; i < 16; i++) cm = fmaxf(cm, red[tid * 16 + i]);
      mnew[tid] = cm;
      rrow[tid] = __expf(mo - cm);
    }
    __syncthreads();
    // --- parallel exp + partial sums, write P~ (bf16) ---
    {
      int r = tid >> 4, seg = (tid & 15) * 16;
      float cm = mnew[r], ls = 0.f;
#pragma unroll
      for (int i = 0; i < 16; i++) {
        float p = __expf(Sb[r * 256 + seg + i] - cm);
        ls += p;
        Pb[r * 256 + seg + i] = (__bf16)p;
      }
      red[r * 16 + (tid & 15)] = ls;
    }
    __syncthreads();
    if (tid < 16) {
      float s = 0.f;
#pragma unroll
      for (int i = 0; i < 16; i++) s += red[tid * 16 + i];
      lrow[tid] = lrow[tid] * rrow[tid] + s;
      mrow[tid] = mnew[tid];
    }
    __syncthreads();
    // --- rescale O then accumulate O += P~ @ x_chunk (this wave's d-slice) ---
#pragma unroll
    for (int j = 0; j < 8; j++) {
      float rs = rrow[j + hi8];
#pragma unroll
      for (int dt = 0; dt < 8; dt++) acc[dt][j] *= rs;
    }
    for (int kk = 0; kk < 256; kk += 32) {
      v16bf a = frag_a(&Pb[lr * 256 + kk], hi8);
#pragma unroll
      for (int dt = 0; dt < 8; dt++) {
        const __bf16* col = xtp + (size_t)(dbase + dt * 16 + lr) * N + n0 + kk;
        acc[dt] = wmma_bf16(a, frag_b(col, hi16), acc[dt]);
      }
    }
    __syncthreads();
  }
  __bf16* To = T + ((size_t)b * 512 + s0) * D;
#pragma unroll
  for (int j = 0; j < 8; j++) {
    float inv = 1.0f / lrow[j + hi8];
    int sr = j + hi8;
#pragma unroll
    for (int dt = 0; dt < 8; dt++)
      To[(size_t)sr * D + dbase + dt * 16 + lr] = (__bf16)(acc[dt][j] * inv);
  }
}

// ---------------------------------------------------------------------------
// K3: out = T @ Wv + bv   ([16384,1024] x [1024,1024], fp32 output)
// Software-pipelined staging (regs hold next chunk during WMMA).
// ---------------------------------------------------------------------------
__global__ void k_outproj(const __bf16* __restrict__ Tin, const float* __restrict__ Wv,
                          const float* __restrict__ bv, float* __restrict__ out) {
  __shared__ __bf16 lA[128 * 40];
  __shared__ __bf16 lB[128 * 40];
  const int D = 1024, P = 1024;
  int p0 = blockIdx.x * 128;
  size_t r0 = (size_t)blockIdx.y * 128;
  int tid = threadIdx.x, w = tid >> 5, lane = tid & 31;
  int hi8 = (lane >> 4) * 8, hi16 = (lane >> 4) * 16, lr = lane & 15;
  int wm = (w >> 2) * 64, wn = (w & 3) * 32;
  v8f acc[4][2] = {};
  int arow = tid >> 1, ac0 = (tid & 1) * 16;
  int bdd = tid & 31, bpp0 = (tid >> 5) * 16;
  const __bf16* paT = Tin + (r0 + arow) * D + ac0;
  const float*  pbW = Wv + (size_t)bdd * P + p0 + bpp0;
  uint4 ra0, ra1; float4 rb[4];
  ra0 = ((const uint4*)paT)[0]; ra1 = ((const uint4*)paT)[1];
#pragma unroll
  for (int i = 0; i < 4; i++) rb[i] = ((const float4*)pbW)[i];
  for (int k0 = 0; k0 < D; k0 += 32) {
    uint4* da = (uint4*)&lA[arow * 40 + ac0]; da[0] = ra0; da[1] = ra1;
    Pack16 u; cvt16(u, rb);
#pragma unroll
    for (int i = 0; i < 16; i++) lB[(bpp0 + i) * 40 + bdd] = u.h[i];  // transpose lB[p][d]
    __syncthreads();
    if (k0 + 32 < D) {                           // overlap next loads with WMMA
      const __bf16* na = paT + k0 + 32;
      const float*  nb = pbW + (size_t)(k0 + 32) * P;
      ra0 = ((const uint4*)na)[0]; ra1 = ((const uint4*)na)[1];
#pragma unroll
      for (int i = 0; i < 4; i++) rb[i] = ((const float4*)nb)[i];
    }
    v16bf bfr[2];
#pragma unroll
    for (int nt = 0; nt < 2; nt++) bfr[nt] = frag_b(&lB[(wn + nt * 16 + lr) * 40], hi16);
#pragma unroll
    for (int mt = 0; mt < 4; mt++) {
      v16bf a = frag_a(&lA[(wm + mt * 16 + lr) * 40], hi8);
#pragma unroll
      for (int nt = 0; nt < 2; nt++) acc[mt][nt] = wmma_bf16(a, bfr[nt], acc[mt][nt]);
    }
    __syncthreads();
  }
#pragma unroll
  for (int mt = 0; mt < 4; mt++)
#pragma unroll
    for (int nt = 0; nt < 2; nt++) {
      float bias = bv[p0 + wn + nt * 16 + lr];
#pragma unroll
      for (int j = 0; j < 8; j++) {
        size_t rr = r0 + wm + mt * 16 + j + hi8;
        out[rr * P + p0 + wn + nt * 16 + lr] = acc[mt][nt][j] + bias;
      }
    }
}

// ---------------------------------------------------------------------------
extern "C" void kernel_launch(void* const* d_in, const int* in_sizes, int n_in,
                              void* d_out, int out_size, void* d_ws, size_t ws_size,
                              hipStream_t stream) {
  const float* x    = (const float*)d_in[0];   // [32,2048,1024]
  const float* seed = (const float*)d_in[1];   // [512,1024]
  const float* Wk   = (const float*)d_in[2];   // [1024,1024]
  // d_in[3] = bk : algebraically eliminated (softmax shift invariance)
  const float* Wv   = (const float*)d_in[4];   // [1024,1024]
  const float* bv   = (const float*)d_in[5];   // [1024]
  float* out = (float*)d_out;                  // [32,512,1024] fp32

  char* ws = (char*)d_ws;
  __bf16* xb = (__bf16*)(ws);                          // 134,217,728 B
  __bf16* xt = (__bf16*)(ws + 134217728ull);           // 134,217,728 B
  __bf16* Qb = (__bf16*)(ws + 268435456ull);           //   1,048,576 B
  __bf16* T  = (__bf16*)(ws + 269484032ull);           //  33,554,432 B

  k_convert<<<dim3(32, 16, 32), 256, 0, stream>>>(x, xb, xt);
  k_qproj  <<<dim3(8, 4),       256, 0, stream>>>(seed, Wk, Qb);
  k_attn   <<<dim3(32, 32),     256, 0, stream>>>(Qb, xb, xt, T);
  k_outproj<<<dim3(8, 128),     256, 0, stream>>>(T, Wv, bv, out);
}
